// MergedConvLiquid_35021163331817
// MI455X (gfx1250) — matrix operand order repaired and checked by reading, not compiled, not run on hardware
//
#include <hip/hip_runtime.h>

#define Bn 4
#define Sn 2048
#define Hn 1024
#define Kc 4
#define HHn 512
#define TWO_H 2048
#define NWG 32
#define NTHR 128
#define DT_C 0.1f
#define TAU_MIN_C 1.0f
#define TAU_MAX_C 5.0f
#define SCALE_C 0.01f
#define LN_EPS_C 1e-5f

typedef __attribute__((ext_vector_type(16))) __bf16 v16bf;
typedef __attribute__((ext_vector_type(8))) float v8f;
typedef __attribute__((ext_vector_type(8))) unsigned int v8u;

__device__ __forceinline__ unsigned short f2bf(float x) {
  unsigned u = __builtin_bit_cast(unsigned, x);
  u += 0x7FFFu + ((u >> 16) & 1u);               // round-to-nearest-even
  return (unsigned short)(u >> 16);
}
__device__ __forceinline__ float bf2f(unsigned short b) {
  return __builtin_bit_cast(float, ((unsigned)b) << 16);
}
// K mapping used here for A and B (contiguous): lane<16 holds K {0..15}, lane>=16 holds {16..31}
__device__ __forceinline__ int kmap(int v, int lane) {
  return 2 * v + ((lane & 16) ? 16 : 0);
}

__device__ __forceinline__ v16bf frag_ld(const unsigned short* p) {
  const uint4* q = (const uint4*)p;
  uint4 a = q[0], b = q[1];
  v8u r = {a.x, a.y, a.z, a.w, b.x, b.y, b.z, b.w};
  return __builtin_bit_cast(v16bf, r);
}
__device__ __forceinline__ v16bf frag_ld_u32(const unsigned* p) {
  return frag_ld((const unsigned short*)p);
}
__device__ __forceinline__ v8f wmma_bf16(v16bf a, v16bf b, v8f c) {
  return __builtin_amdgcn_wmma_f32_16x16x32_bf16(false, a, false, b, (short)0, c, false, false);
}

__device__ __forceinline__ void async_ld_16B(unsigned lds_off, const void* gaddr) {
  asm volatile("global_load_async_to_lds_b128 %0, %1, off"
               :: "v"(lds_off), "v"((unsigned long long)(uintptr_t)gaddr)
               : "memory");
}
__device__ __forceinline__ void wait_async0() {
  asm volatile("s_wait_asynccnt 0" ::: "memory");
}

// ---------------- init ----------------
__global__ void lc_init_kernel(unsigned* bar) {
  if (threadIdx.x == 0 && blockIdx.x == 0) *bar = 0u;
}

// ---------------- pack W_rec into WMMA A-fragment layout, split bf16 hi/lo ----------------
__global__ __launch_bounds__(256) void lc_packW_kernel(const float* __restrict__ W,
                                                       unsigned* __restrict__ hi,
                                                       unsigned* __restrict__ lo) {
  int tid = blockIdx.x * blockDim.x + threadIdx.x;  // 64 mt * 32 kt * 32 lanes = 65536
  int lane = tid & 31, kt = (tid >> 5) & 31, mt = tid >> 10;
  int row = mt * 16 + (lane & 15);
  size_t base = (size_t)tid * 8;
#pragma unroll
  for (int v = 0; v < 8; ++v) {
    int col = kt * 32 + kmap(v, lane);
    float w0 = W[(size_t)row * Hn + col], w1 = W[(size_t)row * Hn + col + 1];
    unsigned short h0 = f2bf(w0), h1 = f2bf(w1);
    unsigned short l0 = f2bf(w0 - bf2f(h0)), l1 = f2bf(w1 - bf2f(h1));
    hi[base + v] = (unsigned)h0 | ((unsigned)h1 << 16);
    lo[base + v] = (unsigned)l0 | ((unsigned)l1 << 16);
  }
}

// ---------------- pack tau_w1^T (A rows = hidden units) ----------------
__global__ __launch_bounds__(256) void lc_packT_kernel(const float* __restrict__ TW1,
                                                       unsigned* __restrict__ hi,
                                                       unsigned* __restrict__ lo) {
  int tid = blockIdx.x * blockDim.x + threadIdx.x;  // 32 mt * 64 kt * 32 lanes = 65536
  int lane = tid & 31, kt = (tid >> 5) & 63, mt = tid >> 11;
  int u = mt * 16 + (lane & 15);
  size_t base = (size_t)tid * 8;
#pragma unroll
  for (int v = 0; v < 8; ++v) {
    int j = kt * 32 + kmap(v, lane);
    float w0 = TW1[(size_t)j * HHn + u], w1 = TW1[(size_t)(j + 1) * HHn + u];
    unsigned short h0 = f2bf(w0), h1 = f2bf(w1);
    unsigned short l0 = f2bf(w0 - bf2f(h0)), l1 = f2bf(w1 - bf2f(h1));
    hi[base + v] = (unsigned)h0 | ((unsigned)h1 << 16);
    lo[base + v] = (unsigned)l0 | ((unsigned)l1 << 16);
  }
}

// ---------------- depthwise causal conv, writes (1-SCALE)*sc ----------------
__global__ __launch_bounds__(256) void lc_conv_kernel(const float* __restrict__ x,
                                                      const float* __restrict__ w,
                                                      float* __restrict__ out) {
  int gid = blockIdx.x * blockDim.x + threadIdx.x;  // B*S*(H/4)
  int c4 = gid & (Hn / 4 - 1);
  int s = (gid >> 8) & (Sn - 1);
  int b = gid >> 19;
  int c = c4 * 4;
  float4 w0 = *(const float4*)(w + (size_t)(c + 0) * Kc);
  float4 w1 = *(const float4*)(w + (size_t)(c + 1) * Kc);
  float4 w2 = *(const float4*)(w + (size_t)(c + 2) * Kc);
  float4 w3 = *(const float4*)(w + (size_t)(c + 3) * Kc);
  float a0 = 0.f, a1 = 0.f, a2 = 0.f, a3 = 0.f;
#pragma unroll
  for (int k = 0; k < Kc; ++k) {
    int sp = s - (Kc - 1) + k;
    if (sp < 0) continue;
    float4 xv = *(const float4*)(x + ((size_t)b * Sn + sp) * Hn + c);
    float wk0 = (&w0.x)[k], wk1 = (&w1.x)[k], wk2 = (&w2.x)[k], wk3 = (&w3.x)[k];
    a0 += wk0 * xv.x; a1 += wk1 * xv.y; a2 += wk2 * xv.z; a3 += wk3 * xv.w;
  }
  float4 o;
  o.x = (1.f - SCALE_C) * a0; o.y = (1.f - SCALE_C) * a1;
  o.z = (1.f - SCALE_C) * a2; o.w = (1.f - SCALE_C) * a3;
  *(float4*)(out + ((size_t)b * Sn + s) * Hn + c) = o;
}

// ---------------- persistent liquid recurrence ----------------
__global__ __launch_bounds__(NTHR, 1) void lc_liquid_kernel(
    const float* __restrict__ x, const float* __restrict__ tb1,
    const float* __restrict__ tw2, const float* __restrict__ tb2,
    const float* __restrict__ lng, const float* __restrict__ lnb,
    const unsigned* __restrict__ Whi, const unsigned* __restrict__ Wlo,
    const unsigned* __restrict__ Thi, const unsigned* __restrict__ Tlo,
    float* __restrict__ fbuf, float* __restrict__ taupart,
    unsigned* __restrict__ bar, float* __restrict__ out) {
  extern __shared__ char smem[];
  unsigned* WAhi = (unsigned*)smem;          // 2 mt * 32 kt * 32 lanes * 8 = 16384 u32
  unsigned* WAlo = WAhi + 16384;
  unsigned* TAhi = WAlo + 16384;             // 64 kt * 32 * 8 = 16384 u32
  unsigned* TAlo = TAhi + 16384;
  unsigned short* comb_hi = (unsigned short*)(TAlo + 16384);  // [4][2048] bf16: [x_t | h]
  unsigned short* comb_lo = comb_hi + Bn * TWO_H;
  float* tauD = (float*)(comb_lo + Bn * TWO_H);  // 2 waves * 32 lanes * 8
  float* red = tauD + 512;                       // 128 * 5 stats partials
  float* scal = red + NTHR * 5;                  // c[4], mu[4], inv[4]
  unsigned short* zpad = (unsigned short*)(scal + 16);  // 16 bf16 zeros (32B, aligned)

  const int wg = blockIdx.x, tid = threadIdx.x;
  const int wave = tid >> 5, lane = tid & 31;
  const int bcol = lane & 15, grp = lane >> 4;
  const bool act = bcol < Bn;

  // One-time: async-copy this WG's weight fragments (~293KB LDS) into LDS
  {
    const unsigned lbW = (unsigned)(uintptr_t)WAhi;
    const unsigned lbWl = (unsigned)(uintptr_t)WAlo;
    const unsigned lbT = (unsigned)(uintptr_t)TAhi;
    const unsigned lbTl = (unsigned)(uintptr_t)TAlo;
    const char* gW = (const char*)(Whi + (size_t)wg * 16384);
    const char* gWl = (const char*)(Wlo + (size_t)wg * 16384);
    const char* gT = (const char*)(Thi + (size_t)wg * 16384);
    const char* gTl = (const char*)(Tlo + (size_t)wg * 16384);
    for (int off = tid * 16; off < 65536; off += NTHR * 16) {
      async_ld_16B(lbW + off, gW + off);
      async_ld_16B(lbWl + off, gWl + off);
      async_ld_16B(lbT + off, gT + off);
      async_ld_16B(lbTl + off, gTl + off);
    }
    for (int e = tid; e < Bn * TWO_H; e += NTHR) { comb_hi[e] = 0; comb_lo[e] = 0; }  // h0 = 0
    if (tid < 16) zpad[tid] = 0;  // zero page for unused WMMA B-columns
    wait_async0();
  }
  __syncthreads();

  const float tb2v = tb2[0];

  for (int t = 0; t < Sn; ++t) {
    const int pb = t & 1;
    // ---- pack x_t into combined (bf16 hi/lo) ----
    {
      int b = tid >> 5, i0 = tid & 31;
#pragma unroll 4
      for (int j = 0; j < 32; ++j) {
        int i = i0 + j * 32;
        float xv = x[((size_t)b * Sn + t) * Hn + i];
        unsigned short h = f2bf(xv);
        comb_hi[b * TWO_H + i] = h;
        comb_lo[b * TWO_H + i] = f2bf(xv - bf2f(h));
      }
    }
    __syncthreads();

    // ---- WMMA phase: waves 0-1 main matmul tiles, waves 2-3 tau MLP K-halves ----
    if (wave < 2) {
      v8f acc = {};
      const unsigned* Ah = WAhi + wave * 32 * 32 * 8;
      const unsigned* Al = WAlo + wave * 32 * 32 * 8;
      for (int kt = 0; kt < 32; ++kt) {
        v16bf a_hi = frag_ld_u32(Ah + ((size_t)kt * 32 + lane) * 8);
        v16bf a_lo = frag_ld_u32(Al + ((size_t)kt * 32 + lane) * 8);
        int idx = bcol * TWO_H + Hn + kt * 32 + grp * 16;
        v16bf b_hi = frag_ld(act ? comb_hi + idx : zpad);  // zero-page select, no branch
        v16bf b_lo = frag_ld(act ? comb_lo + idx : zpad);
        acc = wmma_bf16(a_hi, b_hi, acc);
        acc = wmma_bf16(a_hi, b_lo, acc);
        acc = wmma_bf16(a_lo, b_hi, acc);
      }
      if (act) {
        int ibase = wg * 32 + wave * 16 + grp * 8;
#pragma unroll
        for (int r = 0; r < 8; ++r) {
          int i = ibase + r;
          float f = tanhf(x[((size_t)bcol * Sn + t) * Hn + i] + acc[r]);
          fbuf[(size_t)(pb * Bn + bcol) * Hn + i] = f;
        }
      }
    } else {
      v8f acc = {};
      const int kh = (wave - 2) * 32;
      for (int k2 = 0; k2 < 32; ++k2) {
        int kt = kh + k2;
        v16bf a_hi = frag_ld_u32(TAhi + ((size_t)kt * 32 + lane) * 8);
        v16bf a_lo = frag_ld_u32(TAlo + ((size_t)kt * 32 + lane) * 8);
        int idx = bcol * TWO_H + kt * 32 + grp * 16;
        v16bf b_hi = frag_ld(act ? comb_hi + idx : zpad);
        v16bf b_lo = frag_ld(act ? comb_lo + idx : zpad);
        acc = wmma_bf16(a_hi, b_hi, acc);
        acc = wmma_bf16(a_hi, b_lo, acc);
        acc = wmma_bf16(a_lo, b_hi, acc);
      }
#pragma unroll
      for (int r = 0; r < 8; ++r) tauD[(wave - 2) * 256 + lane * 8 + r] = acc[r];
    }
    __syncthreads();

    // ---- tau partial: tanh of WG's 16 hidden units dotted with tau_w2 slice ----
    if (tid < 8) {
      int b = tid & 3, g2 = tid >> 2;
      int lane2 = g2 * 16 + b;
      float s = 0.f;
#pragma unroll
      for (int r = 0; r < 8; ++r) {
        int u = wg * 16 + g2 * 8 + r;
        float z = tauD[lane2 * 8 + r] + tauD[256 + lane2 * 8 + r] + tb1[u];
        s += tanhf(z) * tw2[u];
      }
      red[tid] = s;
    }
    __syncthreads();
    if (tid < 4) taupart[(size_t)(pb * NWG + wg) * 4 + tid] = red[tid] + red[tid + 4];
    __syncthreads();

    // ---- grid-wide barrier (monotonic counter) ----
    if (tid == 0) {
      __hip_atomic_fetch_add(bar, 1u, __ATOMIC_RELEASE, __HIP_MEMORY_SCOPE_AGENT);
      const unsigned target = (unsigned)(t + 1) * NWG;
      while (__hip_atomic_load(bar, __ATOMIC_ACQUIRE, __HIP_MEMORY_SCOPE_AGENT) < target)
        __builtin_amdgcn_s_sleep(2);
    }
    __syncthreads();

    // ---- stats partials over full H: h from LDS, f from global fbuf ----
    {
      int b = tid >> 5, i0 = tid & 31;
      float sh = 0.f, shh = 0.f, sf = 0.f, sff = 0.f, shf = 0.f;
#pragma unroll 4
      for (int j = 0; j < 32; ++j) {
        int i = i0 + j * 32;
        float h = bf2f(comb_hi[b * TWO_H + Hn + i]) + bf2f(comb_lo[b * TWO_H + Hn + i]);
        float f = fbuf[(size_t)(pb * Bn + b) * Hn + i];
        sh += h; shh += h * h; sf += f; sff += f * f; shf += h * f;
      }
      red[tid * 5 + 0] = sh; red[tid * 5 + 1] = shh; red[tid * 5 + 2] = sf;
      red[tid * 5 + 3] = sff; red[tid * 5 + 4] = shf;
    }
    __syncthreads();

    // ---- tau + closed-form LayerNorm of h' = k*(c*h + f); fixed-order reductions ----
    if (tid < 4) {
      const int b = tid;
      float s = tb2v;
      for (int w = 0; w < NWG; ++w) s += taupart[(size_t)(pb * NWG + w) * 4 + b];
      float tau = TAU_MIN_C + (TAU_MAX_C - TAU_MIN_C) / (1.f + __expf(-s));
      float kk = DT_C / tau, cc = tau / DT_C - 1.f;
      float sh = 0.f, shh = 0.f, sf = 0.f, sff = 0.f, shf = 0.f;
      for (int q = 0; q < 32; ++q) {
        int tt = b * 32 + q;
        sh += red[tt * 5 + 0]; shh += red[tt * 5 + 1]; sf += red[tt * 5 + 2];
        sff += red[tt * 5 + 3]; shf += red[tt * 5 + 4];
      }
      float mu = (cc * sh + sf) * (1.f / Hn);
      float e2 = (cc * cc * shh + 2.f * cc * shf + sff) * (1.f / Hn);
      float var = e2 - mu * mu;
      float inv = kk * rsqrtf(kk * kk * var + LN_EPS_C);  // exact eps handling
      scal[b] = cc; scal[4 + b] = mu; scal[8 + b] = inv;
    }
    __syncthreads();

    // ---- h_next = (u - mu)*inv*g + b; repack bf16 hi/lo; write output slice ----
    {
      int b = tid >> 5, i0 = tid & 31;
      float cc = scal[b], mu = scal[4 + b], inv = scal[8 + b];
#pragma unroll 4
      for (int j = 0; j < 32; ++j) {
        int i = i0 + j * 32;
        float h = bf2f(comb_hi[b * TWO_H + Hn + i]) + bf2f(comb_lo[b * TWO_H + Hn + i]);
        float f = fbuf[(size_t)(pb * Bn + b) * Hn + i];
        float u = cc * h + f;
        float hn = (u - mu) * inv * lng[i] + lnb[i];
        unsigned short hb = f2bf(hn);
        comb_hi[b * TWO_H + Hn + i] = hb;
        comb_lo[b * TWO_H + Hn + i] = f2bf(hn - bf2f(hb));
        if (j == wg) {  // this WG owns output columns [32*wg, 32*wg+32)
          size_t o = ((size_t)b * Sn + t) * Hn + i;
          out[o] += SCALE_C * hn;
        }
      }
    }
    __syncthreads();
  }
}

extern "C" void kernel_launch(void* const* d_in, const int* in_sizes, int n_in,
                              void* d_out, int out_size, void* d_ws, size_t ws_size,
                              hipStream_t stream) {
  const float* x = (const float*)d_in[0];
  const float* convw = (const float*)d_in[1];
  const float* Wrec = (const float*)d_in[2];
  const float* tw1 = (const float*)d_in[3];
  const float* tb1 = (const float*)d_in[4];
  const float* tw2 = (const float*)d_in[5];
  const float* tb2 = (const float*)d_in[6];
  const float* lng = (const float*)d_in[7];
  const float* lnb = (const float*)d_in[8];
  float* out = (float*)d_out;

  char* ws = (char*)d_ws;
  unsigned* Whi = (unsigned*)(ws + (size_t)0);
  unsigned* Wlo = (unsigned*)(ws + ((size_t)2 << 20));
  unsigned* Thi = (unsigned*)(ws + ((size_t)4 << 20));
  unsigned* Tlo = (unsigned*)(ws + ((size_t)6 << 20));
  float* fbuf = (float*)(ws + ((size_t)8 << 20));
  float* taupart = (float*)(ws + ((size_t)8 << 20) + 32768);
  unsigned* bar = (unsigned*)(ws + ((size_t)8 << 20) + 32768 + 4096);

  constexpr int SMEM_BYTES = 16384 * 4 * 4      /* A-fragments */
                           + Bn * TWO_H * 2 * 2 /* comb hi/lo */
                           + 512 * 4 + NTHR * 5 * 4 + 16 * 4
                           + 64 /* zero page */;
  static_assert(SMEM_BYTES <= 320 * 1024, "LDS budget");
  (void)hipFuncSetAttribute((const void*)lc_liquid_kernel,
                            hipFuncAttributeMaxDynamicSharedMemorySize, SMEM_BYTES);

  lc_init_kernel<<<1, 1, 0, stream>>>(bar);
  lc_packW_kernel<<<256, 256, 0, stream>>>(Wrec, Whi, Wlo);
  lc_packT_kernel<<<256, 256, 0, stream>>>(tw1, Thi, Tlo);
  lc_conv_kernel<<<(Bn * Sn * (Hn / 4)) / 256, 256, 0, stream>>>(x, convw, out);
  lc_liquid_kernel<<<NWG, NTHR, SMEM_BYTES, stream>>>(
      x, tb1, tw2, tb2, lng, lnb, Whi, Wlo, Thi, Tlo, fbuf, taupart, bar, out);
}
